// MaskDynamicConv_89670327205903
// MI455X (gfx1250) — matrix operand order repaired and checked by reading, not compiled, not run on hardware
//
#include <hip/hip_runtime.h>
#include <hip/hip_bf16.h>
#include <math.h>

// ---------------- problem constants ----------------
#define Bn   8
#define Ln   100
#define LPn  112      // l padded to 7*16
#define Cn   256
#define HWn  16384
#define NP   32       // pixels per inner tile
#define ITERS 8       // pixel tiles per block
#define BLKS_PER_B 64 // HW / (NP*ITERS)

// ---------------- WMMA helper types ----------------
typedef _Float16 half8  __attribute__((ext_vector_type(8)));
typedef _Float16 half16 __attribute__((ext_vector_type(16)));
typedef float    f32x8  __attribute__((ext_vector_type(8)));
typedef unsigned int u32x4 __attribute__((ext_vector_type(4)));
typedef unsigned int u32x8 __attribute__((ext_vector_type(8)));

union H16 { half16 v; half8 h[2]; };

// A-matrix (16x32 f16): per lane two 16-byte chunks at +0 and +16 halves
__device__ __forceinline__ half16 ldA16(const _Float16* p0) {
  H16 u; u.h[0] = *(const half8*)(p0); u.h[1] = *(const half8*)(p0 + 16); return u.v;
}
// B-matrix (32x16 f16): per lane two 16-byte chunks at +0 and +16 halves
__device__ __forceinline__ half16 ldB16(const _Float16* p0) {
  H16 u; u.h[0] = *(const half8*)(p0); u.h[1] = *(const half8*)(p0 + 16); return u.v;
}
__device__ __forceinline__ f32x8 wmma_f16(half16 a, half16 b, f32x8 c) {
  return __builtin_amdgcn_wmma_f32_16x16x32_f16(false, a, false, b, (short)0, c,
                                                false, false);
}

// ---- Tensor Data Mover: 2D f32 tile (tile_d0 x tile_d1) global -> LDS ----
// D# bit layout per CDNA5 ISA.
__device__ __forceinline__ void tdm_load_2d_f32(const void* gaddr, unsigned lds_off,
                                                unsigned tile_d0, unsigned tile_d1,
                                                unsigned tensor_d0, unsigned tensor_d1,
                                                unsigned stride0) {
  unsigned long long ga = (unsigned long long)(size_t)gaddr;
  u32x4 g0 = { 1u,                                   // count=1, user mode
               lds_off,                               // lds_addr (bytes)
               (unsigned)ga,                          // global_addr[31:0]
               ((unsigned)(ga >> 32) & 0x01FFFFFFu) | (2u << 30) }; // [56:32] | type=2
  u32x8 g1 = { (2u << 16),                            // data_size = 4 bytes
               (tensor_d0 & 0xFFFFu) << 16,           // tensor_dim0[15:0]
               ((tensor_d0 >> 16) & 0xFFFFu) | ((tensor_d1 & 0xFFFFu) << 16),
               ((tensor_d1 >> 16) & 0xFFFFu) | (tile_d0 << 16), // tile_dim0
               (tile_d1 & 0xFFFFu),                   // tile_dim1 (tile_dim2 = 0)
               stride0,                               // tensor_dim0_stride[31:0]
               0u, 0u };
  asm volatile("tensor_load_to_lds %0, %1" :: "s"(g0), "s"(g1) : "memory");
}

// ---------------- workspace layout (bytes) ----------------
#define WS_Q16   0                              // 8*112*256 f16 = 458752
#define WS_WK16  458752                         // 256*256 f16  = 131072
#define WS_WV16  589824                         // 256*256 f16  = 131072
#define WS_PART  720896                         // 8*64*112*256 f32 = 58720256

// =====================================================================
// K0: convert wk, wv -> f16 in ws; zero q16 pad rows (l = 100..111)
// =====================================================================
__global__ __launch_bounds__(256) void mdc_prep(const float* __restrict__ wk,
                                                const float* __restrict__ wv,
                                                _Float16* __restrict__ wk16,
                                                _Float16* __restrict__ wv16,
                                                _Float16* __restrict__ q16) {
  int idx = blockIdx.x * 256 + threadIdx.x;
  if (idx < 65536) {
    wk16[idx] = (_Float16)wk[idx];
  } else if (idx < 131072) {
    int j = idx - 65536;
    wv16[j] = (_Float16)wv[j];
  } else if (idx < 131072 + Bn * 12 * Cn) {
    int j = idx - 131072;            // over 8 * 12 * 256 pad elements
    int b = j / (12 * Cn);
    int r = j % (12 * Cn);
    q16[((size_t)b * LPn + Ln) * Cn + r] = (_Float16)0.0f;
  }
}

// =====================================================================
// K1: q = LN(pro @ wq^T + wq_b), store f16 at [b][l][c] (l < 100)
// =====================================================================
__global__ __launch_bounds__(256) void mdc_q(const float* __restrict__ pro,
                                             const float* __restrict__ wq,
                                             const float* __restrict__ wq_b,
                                             const float* __restrict__ lnq_g,
                                             const float* __restrict__ lnq_b,
                                             _Float16* __restrict__ q16) {
  __shared__ float xs[Cn];
  __shared__ float red[256];
  const int t = threadIdx.x;
  const int b = blockIdx.x / Ln;
  const int l = blockIdx.x % Ln;

  xs[t] = pro[((size_t)b * Ln + l) * Cn + t];
  __syncthreads();

  float acc = wq_b[t];
  const float4* w4 = (const float4*)&wq[(size_t)t * Cn];
  const float4* x4 = (const float4*)xs;
#pragma unroll 4
  for (int i = 0; i < Cn / 4; ++i) {
    float4 wv = w4[i], xv = x4[i];
    acc += wv.x * xv.x + wv.y * xv.y + wv.z * xv.z + wv.w * xv.w;
  }

  red[t] = acc; __syncthreads();
  for (int s = 128; s > 0; s >>= 1) { if (t < s) red[t] += red[t + s]; __syncthreads(); }
  float mu = red[0] / (float)Cn; __syncthreads();
  float d = acc - mu;
  red[t] = d * d; __syncthreads();
  for (int s = 128; s > 0; s >>= 1) { if (t < s) red[t] += red[t + s]; __syncthreads(); }
  float var = red[0] / (float)Cn;
  float rs = rsqrtf(var + 1e-5f);
  float y = d * rs * lnq_g[t] + lnq_b[t];
  q16[((size_t)b * LPn + l) * Cn + t] = (_Float16)y;
}

// =====================================================================
// K2: fused  k/v GEMM + LN + attn GEMM + masked softmax(l) + out GEMM
// grid = 8 * 64 blocks, 256 threads (8 wave32)
// =====================================================================
__global__ __launch_bounds__(256, 1) void mdc_fused(
    const float* __restrict__ f, const float* __restrict__ pos,
    const unsigned char* __restrict__ mask,
    const float* __restrict__ wk_b, const float* __restrict__ wv_b,
    const float* __restrict__ lnk_g, const float* __restrict__ lnk_b,
    const float* __restrict__ lnv_g, const float* __restrict__ lnv_b,
    const _Float16* __restrict__ q16, const _Float16* __restrict__ wk16,
    const _Float16* __restrict__ wv16, float* __restrict__ part) {
  __shared__ _Float16 xk16[NP * Cn];    // (f+p) tile, f16 [pix][c]
  __shared__ _Float16 xv16[NP * Cn];    // f tile,     f16 [pix][c]
  // kraw/vraw double as the TDM staging buffers (phases are barrier-separated):
  //   phase 0: TDM writes f -> kraw, pos -> vraw as f32 [c][pix]
  //   phase 1: conversion reads them, produces xk16/xv16
  //   phase 2: GEMM1/2 overwrite them with pre-LN k/v as f32 [pix][c]
  __shared__ float    kraw[NP * Cn];
  __shared__ float    vraw[NP * Cn];
  __shared__ _Float16 kln [NP * Cn];    // k post-LN f16 [pix][c]   (B of attn GEMM)
  __shared__ _Float16 vlnT[Cn * NP];    // v post-LN f16 [c][pix]   (B of out GEMM)
  __shared__ float    attnS[LPn * NP];  // logits f32 [l][pix]
  __shared__ _Float16 attnp[LPn * NP];  // softmax  f16 [l][pix]    (A of out GEMM)
  __shared__ float sredK[8 * NP], qredK[8 * NP], sredV[8 * NP], qredV[8 * NP];
  __shared__ float muK[NP], rsK[NP], muV[NP], rsV[NP];

  const int t    = threadIdx.x;
  const int lane = t & 31;
  const int w    = t >> 5;          // wave id 0..7
  const int lm   = lane & 15;
  const bool lo  = lane < 16;
  const int b    = blockIdx.x >> 6;
  const int blk  = blockIdx.x & 63;

  const int aoff = lo ? 0 : 8;      // A-matrix per-lane K chunk base
  const int boff = lo ? 0 : 8;      // B-matrix per-lane K chunk base

  const float* fB = f   + (size_t)b * Cn * HWn;
  const float* pB = pos + (size_t)b * Cn * HWn;

  // LDS byte offsets of the staging buffers (flat LDS address = addr[31:0])
  const unsigned lds_f = (unsigned)(size_t)(void*)kraw;
  const unsigned lds_p = (unsigned)(size_t)(void*)vraw;

  f32x8 acc[14];                    // out accumulator: 7 l-tiles x 2 co-tiles
#pragma unroll
  for (int i = 0; i < 14; ++i) acc[i] = (f32x8){};

  for (int it = 0; it < ITERS; ++it) {
    const int pix0 = blk * (NP * ITERS) + it * NP;

    // ---- phase 0: TDM DMA of f / pos tiles (256 rows x 32 pixels, f32) ----
    if (t == 0) {                    // wave 0 issues; EXEC ignored by TDM
      tdm_load_2d_f32(fB + pix0, lds_f, NP, Cn, HWn, Cn, HWn);
      tdm_load_2d_f32(pB + pix0, lds_p, NP, Cn, HWn, Cn, HWn);
      __builtin_amdgcn_s_wait_tensorcnt(0);
    }
    __syncthreads();

    // ---- phase 1: convert staged f32 [c][pix] -> f16 [pix][c] ----
    {
      const int p  = t & 31;
      const int cb = t >> 5;
      for (int c = cb; c < Cn; c += 8) {
        float fv = kraw[c * NP + p];
        float pv = vraw[c * NP + p];
        xk16[p * Cn + c] = (_Float16)(fv + pv);
        xv16[p * Cn + c] = (_Float16)fv;
      }
      if (it + 1 < ITERS) {          // warm L2 for the next TDM tile
        const int gp2 = pix0 + NP + p;
        for (int c = cb; c < Cn; c += 64) {
          __builtin_prefetch(&fB[(size_t)c * HWn + gp2], 0, 3);
          __builtin_prefetch(&pB[(size_t)c * HWn + gp2], 0, 3);
        }
      }
    }
    __syncthreads();

    // ---- phase 2a: GEMM1 kraw = xk @ wk^T  (M=32 pix, N=256, K=256) ----
    {
      f32x8 ck[2][2];
#pragma unroll
      for (int m = 0; m < 2; ++m)
#pragma unroll
        for (int n = 0; n < 2; ++n) ck[m][n] = (f32x8){};
#pragma unroll
      for (int kt = 0; kt < 8; ++kt) {
        const int kb = kt * 32;
        half16 a0 = ldA16(&xk16[(0 * 16 + lm) * Cn + kb + aoff]);
        half16 a1 = ldA16(&xk16[(1 * 16 + lm) * Cn + kb + aoff]);
#pragma unroll
        for (int n = 0; n < 2; ++n) {
          const int co = (2 * w + n) * 16 + lm;
          half16 bk = ldB16(&wk16[(size_t)co * Cn + kb + boff]);
          ck[0][n] = wmma_f16(a0, bk, ck[0][n]);
          ck[1][n] = wmma_f16(a1, bk, ck[1][n]);
        }
      }
#pragma unroll
      for (int m = 0; m < 2; ++m)
#pragma unroll
        for (int n = 0; n < 2; ++n) {
          const int co = (2 * w + n) * 16 + lm;
          const float bias = wk_b[co];
#pragma unroll
          for (int r = 0; r < 8; ++r) {
            const int pr = m * 16 + (lo ? r : r + 8);
            kraw[pr * Cn + co] = ck[m][n][r] + bias;
          }
        }
    }
    // ---- phase 2b: GEMM2 vraw = xv @ wv^T ----
    {
      f32x8 cv[2][2];
#pragma unroll
      for (int m = 0; m < 2; ++m)
#pragma unroll
        for (int n = 0; n < 2; ++n) cv[m][n] = (f32x8){};
#pragma unroll
      for (int kt = 0; kt < 8; ++kt) {
        const int kb = kt * 32;
        half16 a0 = ldA16(&xv16[(0 * 16 + lm) * Cn + kb + aoff]);
        half16 a1 = ldA16(&xv16[(1 * 16 + lm) * Cn + kb + aoff]);
#pragma unroll
        for (int n = 0; n < 2; ++n) {
          const int co = (2 * w + n) * 16 + lm;
          half16 bv = ldB16(&wv16[(size_t)co * Cn + kb + boff]);
          cv[0][n] = wmma_f16(a0, bv, cv[0][n]);
          cv[1][n] = wmma_f16(a1, bv, cv[1][n]);
        }
      }
#pragma unroll
      for (int m = 0; m < 2; ++m)
#pragma unroll
        for (int n = 0; n < 2; ++n) {
          const int co = (2 * w + n) * 16 + lm;
          const float bias = wv_b[co];
#pragma unroll
          for (int r = 0; r < 8; ++r) {
            const int pr = m * 16 + (lo ? r : r + 8);
            vraw[pr * Cn + co] = cv[m][n][r] + bias;
          }
        }
    }
    __syncthreads();

    // ---- phase 3: LN over c for k and v (per pixel) ----
    {
      const int p = t & 31;
      const int g = t >> 5;
      float s1k = 0, s2k = 0, s1v = 0, s2v = 0;
      for (int c = g * 32; c < g * 32 + 32; ++c) {
        float xk = kraw[p * Cn + c]; s1k += xk; s2k += xk * xk;
        float xv = vraw[p * Cn + c]; s1v += xv; s2v += xv * xv;
      }
      sredK[g * 32 + p] = s1k; qredK[g * 32 + p] = s2k;
      sredV[g * 32 + p] = s1v; qredV[g * 32 + p] = s2v;
      __syncthreads();
      if (t < 32) {
        float a1 = 0, a2 = 0, b1 = 0, b2 = 0;
        for (int gg = 0; gg < 8; ++gg) {
          a1 += sredK[gg * 32 + t]; a2 += qredK[gg * 32 + t];
          b1 += sredV[gg * 32 + t]; b2 += qredV[gg * 32 + t];
        }
        float mk = a1 / (float)Cn, mv = b1 / (float)Cn;
        muK[t] = mk; rsK[t] = rsqrtf(a2 / (float)Cn - mk * mk + 1e-5f);
        muV[t] = mv; rsV[t] = rsqrtf(b2 / (float)Cn - mv * mv + 1e-5f);
      }
      __syncthreads();
      const float mk = muK[p], rk = rsK[p], mv = muV[p], rv = rsV[p];
      for (int c = g * 32; c < g * 32 + 32; ++c) {
        float kv = (kraw[p * Cn + c] - mk) * rk * lnk_g[c] + lnk_b[c];
        float vv = (vraw[p * Cn + c] - mv) * rv * lnv_g[c] + lnv_b[c];
        kln [p * Cn + c] = (_Float16)kv;
        vlnT[c * NP + p] = (_Float16)vv;
      }
    }
    __syncthreads();

    // ---- phase 4: GEMM3 attn = q @ k^T  (M=112 l, N=32 pix, K=256) ----
    if (w < 7) {
      f32x8 ca[2]; ca[0] = (f32x8){}; ca[1] = (f32x8){};
#pragma unroll
      for (int kt = 0; kt < 8; ++kt) {
        const int kb = kt * 32;
        half16 aq = ldA16(&q16[((size_t)b * LPn + w * 16 + lm) * Cn + kb + aoff]);
#pragma unroll
        for (int n = 0; n < 2; ++n) {
          half16 bk = ldB16(&kln[(n * 16 + lm) * Cn + kb + boff]);
          ca[n] = wmma_f16(aq, bk, ca[n]);
        }
      }
#pragma unroll
      for (int n = 0; n < 2; ++n)
#pragma unroll
        for (int r = 0; r < 8; ++r) {
          const int l  = w * 16 + (lo ? r : r + 8);
          const int px = n * 16 + lm;
          attnS[l * NP + px] = ca[n][r];
        }
    }
    __syncthreads();

    // ---- phase 5: masked softmax over l (per pixel) ----
    if (t < NP) {
      const int p = t;
      if (!mask[pix0 + p]) {
        for (int l = 0; l < LPn; ++l) attnp[l * NP + p] = (_Float16)0.0f;
      } else {
        float m = -INFINITY;
        for (int l = 0; l < Ln; ++l) m = fmaxf(m, attnS[l * NP + p]);
        float s = 0.0f;
        for (int l = 0; l < Ln; ++l) {
          float e = __expf(attnS[l * NP + p] - m);
          attnS[l * NP + p] = e; s += e;
        }
        const float inv = 1.0f / s;
        for (int l = 0; l < Ln; ++l)
          attnp[l * NP + p] = (_Float16)(attnS[l * NP + p] * inv);
        for (int l = Ln; l < LPn; ++l) attnp[l * NP + p] = (_Float16)0.0f;
      }
    }
    __syncthreads();

    // ---- phase 6: GEMM4 out += attn_p @ v  (M=112 l, N=256 c, K=32 pix) ----
    {
      half16 aT[7];
#pragma unroll
      for (int m = 0; m < 7; ++m)
        aT[m] = ldA16(&attnp[(m * 16 + lm) * NP + aoff]);
#pragma unroll
      for (int n = 0; n < 2; ++n) {
        const int co = (2 * w + n) * 16 + lm;
        half16 bv = ldB16(&vlnT[co * NP + boff]);
#pragma unroll
        for (int m = 0; m < 7; ++m)
          acc[m * 2 + n] = wmma_f16(aT[m], bv, acc[m * 2 + n]);
      }
    }
    __syncthreads();   // protect attnp / vlnT / staging before next tile
  }

  // ---- write per-block partial out (f32, [b][blk][l(112)][c]) ----
  float* pout = part + (size_t)(b * BLKS_PER_B + blk) * LPn * Cn;
#pragma unroll
  for (int m = 0; m < 7; ++m)
#pragma unroll
    for (int n = 0; n < 2; ++n) {
      const int co = (2 * w + n) * 16 + lm;
#pragma unroll
      for (int r = 0; r < 8; ++r) {
        const int l = m * 16 + (lo ? r : r + 8);
        pout[(size_t)l * Cn + co] = acc[m * 2 + n][r];
      }
    }
}

// =====================================================================
// K3: reduce partials over 64 blocks, LN + relu, write d_out (f32)
// =====================================================================
__global__ __launch_bounds__(256) void mdc_finish(const float* __restrict__ part,
                                                  const float* __restrict__ ln1_g,
                                                  const float* __restrict__ ln1_b,
                                                  float* __restrict__ out) {
  __shared__ float red[256];
  const int t = threadIdx.x;
  const int b = blockIdx.x / Ln;
  const int l = blockIdx.x % Ln;

  float acc = 0.0f;
  for (int blk = 0; blk < BLKS_PER_B; ++blk)
    acc += part[((size_t)(b * BLKS_PER_B + blk) * LPn + l) * Cn + t];

  red[t] = acc; __syncthreads();
  for (int s = 128; s > 0; s >>= 1) { if (t < s) red[t] += red[t + s]; __syncthreads(); }
  float mu = red[0] / (float)Cn; __syncthreads();
  float d = acc - mu;
  red[t] = d * d; __syncthreads();
  for (int s = 128; s > 0; s >>= 1) { if (t < s) red[t] += red[t + s]; __syncthreads(); }
  float var = red[0] / (float)Cn;
  float y = d * rsqrtf(var + 1e-5f) * ln1_g[t] + ln1_b[t];
  out[((size_t)b * Ln + l) * Cn + t] = fmaxf(y, 0.0f);
}

// =====================================================================
extern "C" void kernel_launch(void* const* d_in, const int* in_sizes, int n_in,
                              void* d_out, int out_size, void* d_ws, size_t ws_size,
                              hipStream_t stream) {
  const float* pro   = (const float*)d_in[0];
  const float* feat  = (const float*)d_in[1];
  const float* pos   = (const float*)d_in[2];
  const unsigned char* mask = (const unsigned char*)d_in[3];
  const float* wq    = (const float*)d_in[4];
  const float* wq_b  = (const float*)d_in[5];
  const float* wk    = (const float*)d_in[6];
  const float* wk_b  = (const float*)d_in[7];
  const float* wv    = (const float*)d_in[8];
  const float* wv_b  = (const float*)d_in[9];
  const float* lnq_g = (const float*)d_in[10];
  const float* lnq_b = (const float*)d_in[11];
  const float* lnk_g = (const float*)d_in[12];
  const float* lnk_b = (const float*)d_in[13];
  const float* lnv_g = (const float*)d_in[14];
  const float* lnv_b = (const float*)d_in[15];
  const float* ln1_g = (const float*)d_in[16];
  const float* ln1_b = (const float*)d_in[17];
  float* out = (float*)d_out;

  char* ws = (char*)d_ws;
  _Float16* q16  = (_Float16*)(ws + WS_Q16);
  _Float16* wk16 = (_Float16*)(ws + WS_WK16);
  _Float16* wv16 = (_Float16*)(ws + WS_WV16);
  float*    part = (float*)(ws + WS_PART);

  mdc_prep<<<dim3((131072 + Bn * 12 * Cn + 255) / 256), dim3(256), 0, stream>>>(
      wk, wv, wk16, wv16, q16);
  mdc_q<<<dim3(Bn * Ln), dim3(256), 0, stream>>>(pro, wq, wq_b, lnq_g, lnq_b, q16);
  mdc_fused<<<dim3(Bn * BLKS_PER_B), dim3(256), 0, stream>>>(
      feat, pos, mask, wk_b, wv_b, lnk_g, lnk_b, lnv_g, lnv_b, q16, wk16, wv16, part);
  mdc_finish<<<dim3(Bn * Ln), dim3(256), 0, stream>>>(part, ln1_g, ln1_b, out);
}
